// SelfAttention_reduce_sumsampling_18073222382373
// MI455X (gfx1250) — matrix operand order repaired and checked by reading, not compiled, not run on hardware
//
#include <hip/hip_runtime.h>
#include <hip/hip_bf16.h>

// MI455X / gfx1250 flash self-attention.
// Shapes: B=8, N=H*W=4096, C=256, CI=32.
// ws layout: [0,2MB) fy bf16 [B][N][32] ; [2MB,4MB) gx bf16 [B][N][32] ;
//            [4MB,6MB) hxT bf16 [B][32][N] ; [6MB,10MB) attn_g f32 [B][N][32].

typedef __attribute__((ext_vector_type(16))) __bf16 v16bf;
typedef __attribute__((ext_vector_type(8)))  float  v8f;

#define NPOS 4096
#define CCH  256
#define CIN  32

union V16 {
  v16bf v;
  unsigned short h[16];
  unsigned int   u[8];
};

// hardware f32 -> bf16 convert (RNE); clang emits v_cvt_*_bf16_f32 on gfx1250
static __device__ __forceinline__ unsigned short f2bf(float f) {
  return __builtin_bit_cast(unsigned short, (__bf16)f);
}

static __device__ __forceinline__ v8f wmma_bf16(v16bf a, v16bf b, v8f c) {
  return __builtin_amdgcn_wmma_f32_16x16x32_bf16(
      /*neg_a=*/false, a, /*neg_b=*/false, b,
      /*c_mod=*/(short)0, c, /*reuse_a=*/false, /*reuse_b=*/false);
}

// A-matrix (16x32, 16-bit) pair->K mapping. lane holds M = lane&15;
// lanes 0-15 cover K in {0..15}, lanes 16-31 cover K in {16..31}.
static __device__ __forceinline__ int kmapA(int v, int kg) {
  return (kg << 4) + 2 * v;
}

// ---------------------------------------------------------------------------
// Kernel 1: projections fy = y*Wfy, gx = x*Wgx, hx = x*Whx  (bf16 outputs)
// One wave = one 16-row tile. hx written transposed for the attention kernel.
// ---------------------------------------------------------------------------
__global__ void __launch_bounds__(128) proj_kernel(
    const float* __restrict__ x, const float* __restrict__ y,
    const float* __restrict__ Wfy, const float* __restrict__ Wgx,
    const float* __restrict__ Whx,
    unsigned short* __restrict__ fy_ws, unsigned short* __restrict__ gx_ws,
    unsigned short* __restrict__ hxT_ws)
{
  const int lane = threadIdx.x & 31;
  const int wave = threadIdx.x >> 5;
  const int rt   = blockIdx.x * 4 + wave;   // 16-row tile id, 0..2047
  const int m    = lane & 15;
  const int kg   = lane >> 4;
  const int row0 = rt * 16;

  v8f accF[2] = {}, accG[2] = {}, accH[2] = {};

  for (int kc = 0; kc < CCH; kc += 32) {
    // A tiles from y (for fy) and x (for gx/hx), converted to bf16
    V16 ay, ax;
#pragma unroll
    for (int v = 0; v < 8; ++v) {
      const int k = kmapA(v, kg);
      const float2 fy2 = *(const float2*)&y[(size_t)(row0 + m) * CCH + kc + k];
      const float2 fx2 = *(const float2*)&x[(size_t)(row0 + m) * CCH + kc + k];
      ay.h[2 * v] = f2bf(fy2.x); ay.h[2 * v + 1] = f2bf(fy2.y);
      ax.h[2 * v] = f2bf(fx2.x); ax.h[2 * v + 1] = f2bf(fx2.y);
    }
    // B tiles (32x16): lane holds column n = lane&15, K = (lane>>4)*16 + e
#pragma unroll
    for (int nt = 0; nt < 2; ++nt) {
      V16 bf, bg, bh;
#pragma unroll
      for (int e = 0; e < 16; ++e) {
        const int k   = (kg << 4) + e;
        const int col = nt * 16 + m;
        bf.h[e] = f2bf(Wfy[(size_t)(kc + k) * CIN + col]);
        bg.h[e] = f2bf(Wgx[(size_t)(kc + k) * CIN + col]);
        bh.h[e] = f2bf(Whx[(size_t)(kc + k) * CIN + col]);
      }
      accF[nt] = wmma_bf16(ay.v, bf.v, accF[nt]);
      accG[nt] = wmma_bf16(ax.v, bg.v, accG[nt]);
      accH[nt] = wmma_bf16(ax.v, bh.v, accH[nt]);
    }
  }

  // D layout: row m = kg*8 + r, col n = lane&15
#pragma unroll
  for (int nt = 0; nt < 2; ++nt) {
#pragma unroll
    for (int r = 0; r < 8; ++r) {
      const int rr  = row0 + kg * 8 + r;       // flat row in [0, B*N)
      const int col = nt * 16 + m;
      fy_ws[(size_t)rr * CIN + col] = f2bf(accF[nt][r]);
      gx_ws[(size_t)rr * CIN + col] = f2bf(accG[nt][r]);
      const int b  = rr >> 12;
      const int np = rr & (NPOS - 1);
      hxT_ws[((size_t)b * CIN + col) * NPOS + np] = f2bf(accH[nt][r]);
    }
  }
}

// ---------------------------------------------------------------------------
// Kernel 2: flash attention. Block = 8 waves x 16 query rows = 128 queries.
// 64-key chunks staged via global_load_async_to_lds_b128 (ASYNCcnt path).
// Online softmax; never materializes the 4096x4096 score matrix.
// ---------------------------------------------------------------------------
__global__ void __launch_bounds__(256) attn_kernel(
    const unsigned short* __restrict__ fy_ws,
    const unsigned short* __restrict__ gx_ws,
    const unsigned short* __restrict__ hxT_ws,
    float* __restrict__ attn_ws)
{
  __shared__ __align__(64) unsigned short gx_s[64][32];    // keys x ci   (4KB)
  __shared__ __align__(64) unsigned short hxT_s[32][64];   // ci x keys   (4KB)
  __shared__ __align__(64) unsigned short pbuf[8][16][64]; // per-wave P (16KB)

  const int lane = threadIdx.x & 31;
  const int wave = threadIdx.x >> 5;
  const int b    = blockIdx.y;
  const int m    = lane & 15;
  const int kg   = lane >> 4;
  const int q0   = blockIdx.x * 128 + wave * 16;

  // Load A_fy (16 queries x 32 ci) once, A-layout dword gathers from bf16 ws.
  V16 afy;
#pragma unroll
  for (int v = 0; v < 8; ++v) {
    const int k = kmapA(v, kg);
    afy.u[v] = *(const unsigned int*)&fy_ws[(size_t)(b * NPOS + q0 + m) * CIN + k];
  }

  float mrun[8], lrun[8];
  v8f O0 = {}, O1 = {};
#pragma unroll
  for (int r = 0; r < 8; ++r) { mrun[r] = -1e30f; lrun[r] = 0.0f; }

  const unsigned short* gxb  = gx_ws  + (size_t)b * NPOS * CIN;
  const unsigned short* hxTb = hxT_ws + (size_t)b * CIN * NPOS;
  const int t = threadIdx.x;

  // per-thread staging coordinates (16B per thread per buffer)
  const int g_row = t >> 2, g_col = (t & 3) * 8;   // gx_s[64][32]
  const int h_ci  = t >> 3, h_col = (t & 7) * 8;   // hxT_s[32][64]
  const unsigned lds_gx = (unsigned)(uintptr_t)&gx_s[g_row][g_col];
  const unsigned lds_hx = (unsigned)(uintptr_t)&hxT_s[h_ci][h_col];

  for (int kc = 0; kc < NPOS; kc += 64) {
    // Async stage of the 64-key chunk straight into LDS (no VGPR bounce).
    {
      const unsigned goff_gx = ((unsigned)(kc + g_row) * CIN + g_col) * 2u;
      const unsigned goff_hx = ((unsigned)h_ci * NPOS + kc + h_col) * 2u;
      asm volatile("global_load_async_to_lds_b128 %0, %1, %2"
                   :: "v"(lds_gx), "v"(goff_gx), "s"(gxb) : "memory");
      asm volatile("global_load_async_to_lds_b128 %0, %1, %2"
                   :: "v"(lds_hx), "v"(goff_hx), "s"(hxTb) : "memory");
      asm volatile("s_wait_asynccnt 0" ::: "memory");
    }
    __syncthreads();

    // S = fy_tile (16x32ci) x gx_chunkT : four 16-key column tiles.
    // B layout: lane n = m (key), K = kg*16+e (ci) -> contiguous 32B in LDS.
    v8f zero = {};
    v8f S0 = wmma_bf16(afy.v, *(const v16bf*)&gx_s[m][kg * 16], zero);
    v8f S1 = wmma_bf16(afy.v, *(const v16bf*)&gx_s[16 + m][kg * 16], zero);
    v8f S2 = wmma_bf16(afy.v, *(const v16bf*)&gx_s[32 + m][kg * 16], zero);
    v8f S3 = wmma_bf16(afy.v, *(const v16bf*)&gx_s[48 + m][kg * 16], zero);

    // Online softmax (row = 16 lanes of a half-wave; cols across lanes)
    float scale[8];
#pragma unroll
    for (int r = 0; r < 8; ++r) {
      float vmax = fmaxf(fmaxf(S0[r], S1[r]), fmaxf(S2[r], S3[r]));
#pragma unroll
      for (int sh = 1; sh < 16; sh <<= 1)
        vmax = fmaxf(vmax, __shfl_xor(vmax, sh, 32));
      const float mn = fmaxf(mrun[r], vmax);
      const float sc = __expf(mrun[r] - mn);
      const float p0 = __expf(S0[r] - mn);
      const float p1 = __expf(S1[r] - mn);
      const float p2 = __expf(S2[r] - mn);
      const float p3 = __expf(S3[r] - mn);
      float rsum = (p0 + p1) + (p2 + p3);
#pragma unroll
      for (int sh = 1; sh < 16; sh <<= 1)
        rsum += __shfl_xor(rsum, sh, 32);
      lrun[r] = lrun[r] * sc + rsum;
      mrun[r] = mn;
      scale[r] = sc;
      // Route P (D-layout) to per-wave LDS tile: row = kg*8+r, col = key
      const int pr = kg * 8 + r;
      pbuf[wave][pr][m]      = f2bf(p0);
      pbuf[wave][pr][16 + m] = f2bf(p1);
      pbuf[wave][pr][32 + m] = f2bf(p2);
      pbuf[wave][pr][48 + m] = f2bf(p3);
    }
    // wave-internal LDS RAW: wait for DS stores before A-layout gathers
    asm volatile("s_wait_dscnt 0" ::: "memory");

    V16 ap0, ap1;   // P as two A-operands: keys 0-31 and 32-63
#pragma unroll
    for (int v = 0; v < 8; ++v) {
      const int k = kmapA(v, kg);
      ap0.u[v] = *(const unsigned int*)&pbuf[wave][m][k];
      ap1.u[v] = *(const unsigned int*)&pbuf[wave][m][32 + k];
    }

    // O += P (16x64key) x hx_chunk (64key x 32ci) : B tiles from hxT in LDS.
#pragma unroll
    for (int r = 0; r < 8; ++r) { O0[r] *= scale[r]; O1[r] *= scale[r]; }
    O0 = wmma_bf16(ap0.v, *(const v16bf*)&hxT_s[m][kg * 16], O0);
    O0 = wmma_bf16(ap1.v, *(const v16bf*)&hxT_s[m][32 + kg * 16], O0);
    O1 = wmma_bf16(ap0.v, *(const v16bf*)&hxT_s[16 + m][kg * 16], O1);
    O1 = wmma_bf16(ap1.v, *(const v16bf*)&hxT_s[16 + m][32 + kg * 16], O1);

    __syncthreads();
  }

  // Normalize and write attn_g (f32) to workspace
  float* ab = attn_ws + (size_t)b * NPOS * CIN;
#pragma unroll
  for (int r = 0; r < 8; ++r) {
    const int q = q0 + kg * 8 + r;
    const float inv = 1.0f / lrun[r];
    ab[(size_t)q * CIN + m]      = O0[r] * inv;
    ab[(size_t)q * CIN + 16 + m] = O1[r] * inv;
  }
}

// ---------------------------------------------------------------------------
// Kernel 3: out = x + sigma * (attn_g @ W_out). K=32 -> one WMMA per tile.
// ---------------------------------------------------------------------------
__global__ void __launch_bounds__(128) out_kernel(
    const float* __restrict__ x, const float* __restrict__ Wout,
    const float* __restrict__ attn_ws, const float* __restrict__ sigmaPtr,
    float* __restrict__ out)
{
  const int lane = threadIdx.x & 31;
  const int wave = threadIdx.x >> 5;
  const int rt   = blockIdx.x * 4 + wave;
  const int m    = lane & 15;
  const int kg   = lane >> 4;
  const int row0 = rt * 16;
  const float sigma = sigmaPtr[0];

  V16 a;
#pragma unroll
  for (int v = 0; v < 8; ++v) {
    const int k = kmapA(v, kg);
    const float2 g = *(const float2*)&attn_ws[(size_t)(row0 + m) * CIN + k];
    a.h[2 * v] = f2bf(g.x);
    a.h[2 * v + 1] = f2bf(g.y);
  }

  for (int nt = 0; nt < 16; ++nt) {
    V16 bw;
#pragma unroll
    for (int e = 0; e < 16; ++e) {
      const int k = (kg << 4) + e;
      bw.h[e] = f2bf(Wout[(size_t)k * CCH + nt * 16 + m]);
    }
    v8f zero = {};
    const v8f D = wmma_bf16(a.v, bw.v, zero);
#pragma unroll
    for (int r = 0; r < 8; ++r) {
      const size_t idx = (size_t)(row0 + kg * 8 + r) * CCH + nt * 16 + m;
      out[idx] = x[idx] + sigma * D[r];
    }
  }
}

extern "C" void kernel_launch(void* const* d_in, const int* in_sizes, int n_in,
                              void* d_out, int out_size, void* d_ws, size_t ws_size,
                              hipStream_t stream) {
  (void)in_sizes; (void)n_in; (void)out_size; (void)ws_size;
  const float* x     = (const float*)d_in[0];
  const float* y     = (const float*)d_in[1];
  const float* Wfy   = (const float*)d_in[2];
  const float* Wgx   = (const float*)d_in[3];
  const float* Whx   = (const float*)d_in[4];
  const float* Wout  = (const float*)d_in[5];
  const float* sigma = (const float*)d_in[6];
  float* out = (float*)d_out;

  char* ws = (char*)d_ws;
  unsigned short* fy_ws  = (unsigned short*)(ws);
  unsigned short* gx_ws  = (unsigned short*)(ws + (size_t)2 * 1024 * 1024);
  unsigned short* hxT_ws = (unsigned short*)(ws + (size_t)4 * 1024 * 1024);
  float*          ag_ws  = (float*)(ws + (size_t)6 * 1024 * 1024);

  // 2048 row-tiles of 16 rows, 4 waves per block
  proj_kernel<<<512, 128, 0, stream>>>(x, y, Wfy, Wgx, Whx, fy_ws, gx_ws, hxT_ws);
  // 32 query-blocks of 128 per batch, 8 batches
  dim3 g2(32, 8);
  attn_kernel<<<g2, 256, 0, stream>>>(fy_ws, gx_ws, hxT_ws, ag_ws);
  out_kernel<<<512, 128, 0, stream>>>(x, Wout, ag_ws, sigma, out);
}
